// InterpretableMultiHeadAttention_74766790689586
// MI455X (gfx1250) — matrix operand order repaired and checked by reading, not compiled, not run on hardware
//
#include <hip/hip_runtime.h>

typedef __attribute__((ext_vector_type(16))) _Float16 v16h;
typedef __attribute__((ext_vector_type(8)))  _Float16 v8h;
typedef __attribute__((ext_vector_type(8)))  float    v8f;
typedef __attribute__((ext_vector_type(4)))  int      v4i;

#define D_MODEL    1024
#define NUM_HEADS  16
#define D_HEAD     64
#define BATCH      4
#define SEQ        2048

// --------------------------------------------------------------------------
// CDNA5 async global->LDS staging (ASYNCcnt path), guarded by __has_builtin.
// --------------------------------------------------------------------------
#if defined(__has_builtin)
#if __has_builtin(__builtin_amdgcn_global_load_async_to_lds_b128)
#define USE_ASYNC_LDS 1
#endif
#endif
#ifndef USE_ASYNC_LDS
#define USE_ASYNC_LDS 0
#endif

static __device__ __forceinline__ void wait_async_le8() {
#if defined(__has_builtin) && __has_builtin(__builtin_amdgcn_s_wait_asynccnt)
  __builtin_amdgcn_s_wait_asynccnt(8);
#else
  asm volatile("s_wait_asynccnt 0x8" ::: "memory");
#endif
}

#if USE_ASYNC_LDS
static __device__ __forceinline__ void async_copy16(const _Float16* g,
                                                    _Float16* lds) {
  __builtin_amdgcn_global_load_async_to_lds_b128(
      (__attribute__((address_space(1))) v4i*)g,
      (__attribute__((address_space(3))) v4i*)lds, 0, 0);
}
#endif

// ---------------------------------------------------------------------------
// WMMA helpers (CDNA5: D = A(16x32 f16) x B(32x16 f16) + C(16x16 f32))
// ---------------------------------------------------------------------------
static __device__ __forceinline__ v8f wmma16(v16h a, v16h b, v8f c) {
  return __builtin_amdgcn_wmma_f32_16x16x32_f16(false, a, false, b, (short)0, c,
                                                false, false);
}

// A fragment from row-major f16: lane (m = l&15, hh = l>>4) holds
// contiguous K = kb+16hh .. +16   (matches B fragment order)
static __device__ __forceinline__ v16h load_a_h(const _Float16* base, int ld, int kb) {
  const unsigned l = threadIdx.x & 31u, m = l & 15u, hh = l >> 4;
  const _Float16* p0 = base + (int)m * ld + kb + 16 * (int)hh;
  v8h lo = *(const v8h*)p0;
  v8h hi = *(const v8h*)(p0 + 8);
  v16h a;
#pragma unroll
  for (int i = 0; i < 8; ++i) { a[i] = lo[i]; a[i + 8] = hi[i]; }
  return a;
}

// A fragment from row-major f32 (convert inline to f16)
static __device__ __forceinline__ v16h load_a_f(const float* base, int ld, int kb) {
  const unsigned l = threadIdx.x & 31u, m = l & 15u, hh = l >> 4;
  const float* p0 = base + (int)m * ld + kb + 16 * (int)hh;
  const float* p1 = p0 + 8;
  v16h a;
#pragma unroll
  for (int i = 0; i < 8; ++i) { a[i] = (_Float16)p0[i]; a[i + 8] = (_Float16)p1[i]; }
  return a;
}

// B fragment (32x16): column n of B = row n of the stored matrix (stride ld);
// lane (n = l&15, hh = l>>4) holds contiguous K = kb+16hh .. +16.
static __device__ __forceinline__ v16h load_b_h(const _Float16* base, int ld, int kb) {
  const unsigned l = threadIdx.x & 31u, n = l & 15u, hh = l >> 4;
  return *(const v16h*)(base + (int)n * ld + kb + 16 * (int)hh);
}

// reductions across the 16-lane half that owns one C-row
static __device__ __forceinline__ float half_max(float x) {
#pragma unroll
  for (int mask = 1; mask <= 8; mask <<= 1) x = fmaxf(x, __shfl_xor(x, mask, 32));
  return x;
}
static __device__ __forceinline__ float half_sum(float x) {
#pragma unroll
  for (int mask = 1; mask <= 8; mask <<= 1) x += __shfl_xor(x, mask, 32);
  return x;
}

// ---------------------------------------------------------------------------
// f32 -> f16 convert (weights)
// ---------------------------------------------------------------------------
__global__ void __launch_bounds__(256)
cvt_kernel(const float* __restrict__ in, _Float16* __restrict__ out, int nElem) {
  for (int i = blockIdx.x * blockDim.x + threadIdx.x; i < nElem;
       i += gridDim.x * blockDim.x)
    out[i] = (_Float16)in[i];
}

// ---------------------------------------------------------------------------
// Projection GEMM: Y = X @ W^T + b   (X f32 [8192,1024], W f16 [1024,1024])
// VMODE 0: store f16 as [B,H,S,64]   (Q, K)
// VMODE 1: store f16 as [B,H,64,S]   (V transposed for PV B-fragments)
// One wave computes a 16x64 tile. B slabs (64x32) staged into LDS by the
// async-copy engine (double buffered, s_wait_asynccnt); A double-buffered in
// registers; all 4 B fragments ds-loaded before the 4 WMMAs so the DS pipe
// runs ahead of the matrix pipe (partial dscnt waits).
// ---------------------------------------------------------------------------
template <int VMODE>
__global__ void __launch_bounds__(128)
proj_qkv_kernel(const float* __restrict__ X, const _Float16* __restrict__ Wh,
                const float* __restrict__ bias, _Float16* __restrict__ Out) {
  const int wave = threadIdx.x >> 5;
  const unsigned l = threadIdx.x & 31u, n = l & 15u, hh = l >> 4;
  const int m0 = blockIdx.x * 16;
  const int n0 = (blockIdx.y * 4 + wave) * 64;

  v8f zero = {};
  v8f acc[4];
#pragma unroll
  for (int f = 0; f < 4; ++f) acc[f] = zero;

  const float* Xrow = X + (size_t)m0 * D_MODEL;

#if USE_ASYNC_LDS
  __shared__ _Float16 sB[4][2][64 * 32];
  auto issue = [&](int kb, int buf) {
#pragma unroll
    for (int i = 0; i < 8; ++i) {
      const int c = i * 32 + (int)l;   // 16-byte chunk id 0..255
      const int r = c >> 2;            // output column 0..63
      const int o = (c & 3) * 8;       // f16 offset within 32-wide k slab
      async_copy16(Wh + (size_t)(n0 + r) * D_MODEL + kb + o,
                   &sB[wave][buf][r * 32 + o]);
    }
  };

  issue(0, 0);
  v16h a_nxt = load_a_f(Xrow, D_MODEL, 0);
  for (int it = 0; it < 32; ++it) {
    const int cur = it & 1;
    const int kn = ((it + 1) & 31) * 32;   // wrapped: last prefetch is harmless
    issue(kn, cur ^ 1);
    const v16h a = a_nxt;
    a_nxt = load_a_f(Xrow, D_MODEL, kn);
    wait_async_le8();                      // in-order: current slab complete
    v16h bf[4];
#pragma unroll
    for (int f = 0; f < 4; ++f)
      bf[f] = load_b_h(&sB[wave][cur][f * 16 * 32], 32, 0);
#pragma unroll
    for (int f = 0; f < 4; ++f) acc[f] = wmma16(a, bf[f], acc[f]);
  }
#else
  v16h a_nxt = load_a_f(Xrow, D_MODEL, 0);
  v16h b_nxt[4];
#pragma unroll
  for (int f = 0; f < 4; ++f)
    b_nxt[f] = load_b_h(Wh + (size_t)(n0 + f * 16) * D_MODEL, D_MODEL, 0);
  for (int it = 0; it < 32; ++it) {
    const int kn = ((it + 1) & 31) * 32;
    const v16h a = a_nxt;
    v16h bf[4];
#pragma unroll
    for (int f = 0; f < 4; ++f) bf[f] = b_nxt[f];
    a_nxt = load_a_f(Xrow, D_MODEL, kn);
#pragma unroll
    for (int f = 0; f < 4; ++f)
      b_nxt[f] = load_b_h(Wh + (size_t)(n0 + f * 16) * D_MODEL, D_MODEL, kn);
#pragma unroll
    for (int f = 0; f < 4; ++f) acc[f] = wmma16(a, bf[f], acc[f]);
  }
#endif

#pragma unroll
  for (int f = 0; f < 4; ++f) {
    const int col = n0 + f * 16 + (int)n;
    const float bb = bias[col];
    const int h = col >> 6, d = col & 63;
#pragma unroll
    for (int v = 0; v < 8; ++v) {
      const int gm = m0 + v + 8 * (int)hh;
      const int b_ = gm >> 11, s_ = gm & (SEQ - 1);
      const float val = acc[f][v] + bb;
      if (VMODE == 0)
        Out[(((size_t)(b_ * NUM_HEADS + h) * SEQ) + s_) * D_HEAD + d] = (_Float16)val;
      else
        Out[(((size_t)(b_ * NUM_HEADS + h) * D_HEAD) + d) * SEQ + s_] = (_Float16)val;
    }
  }
}

// ---------------------------------------------------------------------------
// Output GEMM: out = ctx(f16) @ Wo^T + bo -> f32 flat [8192,1024]
// ---------------------------------------------------------------------------
__global__ void __launch_bounds__(128)
proj_out_kernel(const _Float16* __restrict__ Xh, const _Float16* __restrict__ Wh,
                const float* __restrict__ bias, float* __restrict__ Out) {
  const int wave = threadIdx.x >> 5;
  const unsigned l = threadIdx.x & 31u, n = l & 15u, hh = l >> 4;
  const int m0 = blockIdx.x * 16;
  const int n0 = (blockIdx.y * 4 + wave) * 64;

  v8f zero = {};
  v8f acc[4];
#pragma unroll
  for (int f = 0; f < 4; ++f) acc[f] = zero;

  const _Float16* Xrow = Xh + (size_t)m0 * D_MODEL;

#if USE_ASYNC_LDS
  __shared__ _Float16 sB[4][2][64 * 32];
  auto issue = [&](int kb, int buf) {
#pragma unroll
    for (int i = 0; i < 8; ++i) {
      const int c = i * 32 + (int)l;
      const int r = c >> 2;
      const int o = (c & 3) * 8;
      async_copy16(Wh + (size_t)(n0 + r) * D_MODEL + kb + o,
                   &sB[wave][buf][r * 32 + o]);
    }
  };

  issue(0, 0);
  v16h a_nxt = load_a_h(Xrow, D_MODEL, 0);
  for (int it = 0; it < 32; ++it) {
    const int cur = it & 1;
    const int kn = ((it + 1) & 31) * 32;
    issue(kn, cur ^ 1);
    const v16h a = a_nxt;
    a_nxt = load_a_h(Xrow, D_MODEL, kn);
    wait_async_le8();
    v16h bf[4];
#pragma unroll
    for (int f = 0; f < 4; ++f)
      bf[f] = load_b_h(&sB[wave][cur][f * 16 * 32], 32, 0);
#pragma unroll
    for (int f = 0; f < 4; ++f) acc[f] = wmma16(a, bf[f], acc[f]);
  }
#else
  v16h a_nxt = load_a_h(Xrow, D_MODEL, 0);
  v16h b_nxt[4];
#pragma unroll
  for (int f = 0; f < 4; ++f)
    b_nxt[f] = load_b_h(Wh + (size_t)(n0 + f * 16) * D_MODEL, D_MODEL, 0);
  for (int it = 0; it < 32; ++it) {
    const int kn = ((it + 1) & 31) * 32;
    const v16h a = a_nxt;
    v16h bf[4];
#pragma unroll
    for (int f = 0; f < 4; ++f) bf[f] = b_nxt[f];
    a_nxt = load_a_h(Xrow, D_MODEL, kn);
#pragma unroll
    for (int f = 0; f < 4; ++f)
      b_nxt[f] = load_b_h(Wh + (size_t)(n0 + f * 16) * D_MODEL, D_MODEL, kn);
#pragma unroll
    for (int f = 0; f < 4; ++f) acc[f] = wmma16(a, bf[f], acc[f]);
  }
#endif

#pragma unroll
  for (int f = 0; f < 4; ++f) {
    const int col = n0 + f * 16 + (int)n;
    const float bb = bias[col];
#pragma unroll
    for (int v = 0; v < 8; ++v) {
      const int gm = m0 + v + 8 * (int)hh;
      Out[(size_t)gm * D_MODEL + col] = acc[f][v] + bb;
    }
  }
}

// ---------------------------------------------------------------------------
// Attention: one workgroup (8 waves) per (batch, 16-query tile).
// Each wave handles 2 heads. Two-pass softmax (1/sqrt(Dh) folded into Q);
// K and V fragments double-buffered across tiles; avg-attention accumulated
// in dynamic LDS via ds_add_f32; P reshaped C->A layout through per-wave LDS.
// ---------------------------------------------------------------------------
__global__ void __launch_bounds__(256)
attn_kernel(const _Float16* __restrict__ Qh, const _Float16* __restrict__ Kh,
            const _Float16* __restrict__ Vt, _Float16* __restrict__ Ctx,
            float* __restrict__ Avg) {
  extern __shared__ char smem[];
  float* sAvg = (float*)smem;                                  // 16*SEQ f32
  _Float16* sP = (_Float16*)(smem + (size_t)16 * SEQ * sizeof(float));

  const int tid = threadIdx.x;
  const int wave = tid >> 5;
  const unsigned l = tid & 31u, n = l & 15u, hh = l >> 4;
  const int b = blockIdx.x >> 7;
  const int qbase = (blockIdx.x & 127) * 16;
  const int NT = SEQ / 16;

  for (int i = tid; i < 16 * SEQ; i += 256) sAvg[i] = 0.0f;
  __syncthreads();

  _Float16* myP = sP + wave * (16 * 32);
  v8f zero = {};

  for (int iter = 0; iter < 2; ++iter) {
    const int h = wave + 8 * iter;
    const _Float16* Qb = Qh + ((size_t)(b * NUM_HEADS + h) * SEQ + qbase) * D_HEAD;
    const _Float16* Kb = Kh + (size_t)(b * NUM_HEADS + h) * SEQ * D_HEAD;
    const _Float16* Vb = Vt + (size_t)(b * NUM_HEADS + h) * D_HEAD * SEQ;

    v16h aq0 = load_a_h(Qb, D_HEAD, 0);
    v16h aq1 = load_a_h(Qb, D_HEAD, 32);
#pragma unroll
    for (int i = 0; i < 16; ++i) {       // fold 1/sqrt(64) into Q (exact)
      aq0[i] = aq0[i] * (_Float16)0.125f;
      aq1[i] = aq1[i] * (_Float16)0.125f;
    }

    float rm[8], rl[8];
#pragma unroll
    for (int v = 0; v < 8; ++v) { rm[v] = -__builtin_inff(); rl[v] = 0.0f; }

    // ---- pass 1: row max + sum(exp), pipelined K tiles ----
    v16h pb0 = load_b_h(Kb, D_HEAD, 0);
    v16h pb1 = load_b_h(Kb, D_HEAD, 32);
    for (int j = 0; j < NT; ++j) {
      const v16h b0 = pb0, b1 = pb1;
      const _Float16* Ktn = Kb + (size_t)((j + 1) & (NT - 1)) * 16 * D_HEAD;
      pb0 = load_b_h(Ktn, D_HEAD, 0);
      pb1 = load_b_h(Ktn, D_HEAD, 32);
      v8f s = wmma16(aq0, b0, zero);
      s = wmma16(aq1, b1, s);
#pragma unroll
      for (int v = 0; v < 8; ++v) {
        const float nm = fmaxf(rm[v], half_max(s[v]));
        const float ts = half_sum(__expf(s[v] - nm));
        rl[v] = rl[v] * __expf(rm[v] - nm) + ts;
        rm[v] = nm;
      }
    }
    float rinv[8];
#pragma unroll
    for (int v = 0; v < 8; ++v) rinv[v] = 1.0f / rl[v];

    // ---- pass 2: normalized P, avg accumulation, O += P @ V ----
    v8f o[4];
#pragma unroll
    for (int f = 0; f < 4; ++f) o[f] = zero;

    pb0 = load_b_h(Kb, D_HEAD, 0);
    pb1 = load_b_h(Kb, D_HEAD, 32);
    v16h pv[4];
#pragma unroll
    for (int f = 0; f < 4; ++f)          // prefetch first V slab fragments
      pv[f] = load_b_h(Vb + (size_t)f * 16 * SEQ, SEQ, 0);

    for (int kt = 0; kt < NT; ++kt) {
      const v16h b0 = pb0, b1 = pb1;
      const _Float16* Ktn = Kb + (size_t)((kt + 1) & (NT - 1)) * 16 * D_HEAD;
      pb0 = load_b_h(Ktn, D_HEAD, 0);
      pb1 = load_b_h(Ktn, D_HEAD, 32);
      v8f s = wmma16(aq0, b0, zero);
      s = wmma16(aq1, b1, s);
      const int t = kt & 1;
#pragma unroll
      for (int v = 0; v < 8; ++v) {
        const int row = v + 8 * (int)hh;
        const float p = __expf(s[v] - rm[v]) * rinv[v];
        atomicAdd(&sAvg[row * SEQ + kt * 16 + (int)n], p * (1.0f / NUM_HEADS));
        myP[row * 32 + t * 16 + (int)n] = (_Float16)p;
      }
      if (t) {                                // 32-wide P slab complete
        v16h cv[4];
#pragma unroll
        for (int f = 0; f < 4; ++f) cv[f] = pv[f];
        const int knext = (kt + 1) * 16 & (SEQ - 1);  // next slab (wrapped)
#pragma unroll
        for (int f = 0; f < 4; ++f)
          pv[f] = load_b_h(Vb + (size_t)f * 16 * SEQ, SEQ, knext);
        const v16h ap = load_a_h(myP, 32, 0); // P (16x32) as A fragment
#pragma unroll
        for (int f = 0; f < 4; ++f) o[f] = wmma16(ap, cv[f], o[f]);
      }
    }

    // ---- store context rows: ctx[b, q, h*64 + d] (f16) ----
#pragma unroll
    for (int f = 0; f < 4; ++f)
#pragma unroll
      for (int v = 0; v < 8; ++v) {
        const int row = v + 8 * (int)hh;
        Ctx[(size_t)(b * SEQ + qbase + row) * D_MODEL + h * D_HEAD + f * 16 + (int)n] =
            (_Float16)o[f][v];
      }
  }

  __syncthreads();
  for (int i = tid; i < 16 * SEQ; i += 256) {
    const int row = i >> 11, col = i & (SEQ - 1);
    Avg[(size_t)(b * SEQ + qbase + row) * SEQ + col] = sAvg[i];
  }
}

// ---------------------------------------------------------------------------
// Host launcher
// ---------------------------------------------------------------------------
extern "C" void kernel_launch(void* const* d_in, const int* in_sizes, int n_in,
                              void* d_out, int out_size, void* d_ws, size_t ws_size,
                              hipStream_t stream) {
  const float* q  = (const float*)d_in[0];
  const float* k  = (const float*)d_in[1];
  const float* v  = (const float*)d_in[2];
  const float* Wq = (const float*)d_in[3];
  const float* bq = (const float*)d_in[4];
  const float* Wk = (const float*)d_in[5];
  const float* bk = (const float*)d_in[6];
  const float* Wv = (const float*)d_in[7];
  const float* bv = (const float*)d_in[8];
  const float* Wo = (const float*)d_in[9];
  const float* bo = (const float*)d_in[10];

  const size_t NW = (size_t)D_MODEL * D_MODEL;                    // 1M
  const size_t NP = (size_t)BATCH * NUM_HEADS * SEQ * D_HEAD;     // 8M

  _Float16* ws  = (_Float16*)d_ws;
  _Float16* WqH = ws;
  _Float16* WkH = WqH + NW;
  _Float16* WvH = WkH + NW;
  _Float16* WoH = WvH + NW;
  _Float16* Qh  = WoH + NW;
  _Float16* Kh  = Qh + NP;
  _Float16* Vt  = Kh + NP;
  _Float16* Ctx = Vt + NP;

  float* out = (float*)d_out;
  float* avg = out + (size_t)BATCH * SEQ * D_MODEL;

  cvt_kernel<<<dim3(1024), dim3(256), 0, stream>>>(Wq, WqH, (int)NW);
  cvt_kernel<<<dim3(1024), dim3(256), 0, stream>>>(Wk, WkH, (int)NW);
  cvt_kernel<<<dim3(1024), dim3(256), 0, stream>>>(Wv, WvH, (int)NW);
  cvt_kernel<<<dim3(1024), dim3(256), 0, stream>>>(Wo, WoH, (int)NW);

  dim3 g(512, 4), blk(128);
  proj_qkv_kernel<0><<<g, blk, 0, stream>>>(q, WqH, bq, Qh);
  proj_qkv_kernel<0><<<g, blk, 0, stream>>>(k, WkH, bk, Kh);
  proj_qkv_kernel<1><<<g, blk, 0, stream>>>(v, WvH, bv, Vt);

  const size_t sh = (size_t)16 * SEQ * sizeof(float)
                  + (size_t)8 * 16 * 32 * sizeof(_Float16);       // 136 KB dyn LDS
  attn_kernel<<<dim3(BATCH * (SEQ / 16)), dim3(256), sh, stream>>>(Qh, Kh, Vt,
                                                                   Ctx, avg);

  proj_out_kernel<<<g, blk, 0, stream>>>(Ctx, WoH, bo, out);
}
